// CausalSelfAttention_11269994184974
// MI455X (gfx1250) — matrix operand order repaired and checked by reading, not compiled, not run on hardware
//
#include <hip/hip_runtime.h>

// ---------------------------------------------------------------------------
// Problem constants (from reference): B=4, T=1024, D=1024, H=16, DH=64
// ---------------------------------------------------------------------------
#define BB   4
#define TT   1024
#define DD   1024
#define HH_  16
#define DH_  64
#define NTOK (BB * TT)        // 4096
#define NQKV (3 * HH_ * DH_)  // 3072
#define ATTN_SCALE 0.12f

typedef __bf16 bf16;
typedef __attribute__((ext_vector_type(16))) __bf16 bf16x16;
typedef __attribute__((ext_vector_type(8)))  __bf16 bf16x8;
typedef __attribute__((ext_vector_type(8)))  float  f32x8;

union BFrag { bf16x16 v; bf16x8 h[2]; };

static __device__ __forceinline__ f32x8 wmma_bf16(bf16x16 a, bf16x16 b, f32x8 c) {
  return __builtin_amdgcn_wmma_f32_16x16x32_bf16(false, a, false, b, (short)0, c,
                                                 false, false);
}
static __device__ __forceinline__ f32x8 zero8() {
  f32x8 z; for (int i = 0; i < 8; ++i) z[i] = 0.0f; return z;
}

// CDNA5 async global->LDS copy (16 bytes per lane), tracked by ASYNCcnt.
// LDS operand is the DS-relative byte address = low 32 bits of the generic ptr.
static __device__ __forceinline__ void async_copy_b128(const void* gptr, void* lptr) {
  unsigned lds = (unsigned)(unsigned long long)(uintptr_t)lptr;
  unsigned long long g = (unsigned long long)(uintptr_t)gptr;
  asm volatile("global_load_async_to_lds_b128 %0, %1, off"
               :: "v"(lds), "v"(g) : "memory");
}
static __device__ __forceinline__ void wait_async0() {
  asm volatile("s_wait_asynccnt 0x0" ::: "memory");
}

// ---------------------------------------------------------------------------
// f32 -> bf16 convert (grid-stride)
// ---------------------------------------------------------------------------
__global__ void cvt_f32_bf16(const float* __restrict__ s, bf16* __restrict__ d, int n) {
  int i = blockIdx.x * blockDim.x + threadIdx.x;
  int stride = gridDim.x * blockDim.x;
  for (; i < n; i += stride) d[i] = (bf16)s[i];
}

// ---------------------------------------------------------------------------
// Tiled WMMA GEMM:  C[M,N](f32) = A[M,K](bf16) * B[N,K](bf16)^T
// Block tile 128(M) x 64(N), K-step 32. 8 waves, wave w owns rows 16w..16w+15.
// Double-buffered LDS filled by GLOBAL_LOAD_ASYNC_TO_LDS_B128 so the async
// DMA of tile i+1 overlaps the WMMA phase of tile i.
// ---------------------------------------------------------------------------
__global__ __launch_bounds__(256) void gemm_bf16_nt(
    const bf16* __restrict__ A, const bf16* __restrict__ B,
    float* __restrict__ C, int M, int N, int K) {
  __shared__ bf16 As[2][128 * 32];
  __shared__ bf16 Bs[2][64 * 32];

  const int tid  = threadIdx.x;
  const int wave = tid >> 5;
  const int lane = tid & 31;
  const int m16  = lane & 15;   // fragment row / col within 16
  const int hh   = lane >> 4;   // lane-half selector
  const int rowBase = blockIdx.y * 128;
  const int colBase = blockIdx.x * 64;

  f32x8 acc[4];
  for (int fb = 0; fb < 4; ++fb) acc[fb] = zero8();

  const int aRow = tid >> 1;          // 0..127, 32B (16 bf16) per thread
  const int aCol = (tid & 1) * 16;    // 0 / 16
  const int bRow = tid >> 2;          // 0..63, 16B (8 bf16) per thread
  const int bCol = (tid & 3) * 8;     // 0..24

  const bf16* aG = A + (size_t)(rowBase + aRow) * K + aCol;
  const bf16* bG = B + (size_t)(colBase + bRow) * K + bCol;

  auto issue_tile = [&](int k0, int buf) {
    async_copy_b128(aG + k0,     &As[buf][aRow * 32 + aCol]);
    async_copy_b128(aG + k0 + 8, &As[buf][aRow * 32 + aCol + 8]);
    async_copy_b128(bG + k0,     &Bs[buf][bRow * 32 + bCol]);
  };

  const int steps = K >> 5;   // K / 32
  issue_tile(0, 0);

  for (int i = 0; i < steps; ++i) {
    const int cur = i & 1;
    wait_async0();       // own async loads of tile i have landed in LDS
    __syncthreads();     // everyone's tile-i data visible; tile i-1 consumers done
    if (i + 1 < steps) issue_tile((i + 1) << 5, 1 - cur);

    // A fragment: 16x32, lane m16 = row, runs at K = 8*hh and 16+8*hh
    BFrag a;
    const bf16* arow = &As[cur][(wave * 16 + m16) * 32];
    a.h[0] = *(const bf16x8*)(arow + 8 * hh);
    a.h[1] = *(const bf16x8*)(arow + 16 + 8 * hh);

    for (int fb = 0; fb < 4; ++fb) {
      // B fragment: lane m16 = N-col, runs at K = 8*hh and 16+8*hh
      BFrag b;
      const bf16* brow = &Bs[cur][(fb * 16 + m16) * 32];
      b.h[0] = *(const bf16x8*)(brow + 8 * hh);
      b.h[1] = *(const bf16x8*)(brow + 16 + 8 * hh);
      acc[fb] = wmma_bf16(a.v, b.v, acc[fb]);
    }
  }

  for (int fb = 0; fb < 4; ++fb)
    for (int r = 0; r < 8; ++r) {
      int m = rowBase + wave * 16 + r + 8 * hh;
      int n = colBase + fb * 16 + m16;
      C[(size_t)m * N + n] = acc[fb][r];
    }
}

// ---------------------------------------------------------------------------
// Fused epilogue on qkv: RMS-norm(q,k), rotary(q,k), v-blend with value
// embeds. One wave per (b,t,h) row; lane = d in [0,32), also handles d+32.
// Outputs: qb,kb as [bh][t][dh] bf16; v transposed as vT[bh][dh][t] bf16.
// ---------------------------------------------------------------------------
__global__ __launch_bounds__(256) void fuse_norm_rope_v(
    const float* __restrict__ qkv, const float* __restrict__ ve,
    const float* __restrict__ lam,
    bf16* __restrict__ qb, bf16* __restrict__ kb, bf16* __restrict__ vT) {
  const int R    = blockIdx.x * 8 + (threadIdx.x >> 5);  // row in [0, B*T*H)
  const int lane = threadIdx.x & 31;
  const int b    = R >> 14;          // / (T*H) = 16384
  const int rem  = R & 16383;
  const int t    = rem >> 4;         // / H
  const int h    = rem & 15;
  const int d    = lane;             // 0..31

  const size_t rowOff = ((size_t)(b * TT + t)) * NQKV + h * DH_;
  float q0 = qkv[rowOff + d],        q1 = qkv[rowOff + d + 32];
  float k0 = qkv[rowOff + 1024 + d], k1 = qkv[rowOff + 1024 + d + 32];
  float v0 = qkv[rowOff + 2048 + d], v1 = qkv[rowOff + 2048 + d + 32];

  float ssq = q0 * q0 + q1 * q1;
  float ssk = k0 * k0 + k1 * k1;
  for (int m = 16; m >= 1; m >>= 1) {
    ssq += __shfl_xor(ssq, m);
    ssk += __shfl_xor(ssk, m);
  }
  float iq = rsqrtf(ssq * (1.0f / 64.0f) + 1e-6f);
  float ik = rsqrtf(ssk * (1.0f / 64.0f) + 1e-6f);
  q0 *= iq; q1 *= iq; k0 *= ik; k1 *= ik;

  // rotary: freq = (1/1024)^linspace(0,1,16) for d<16, else 0
  float fi = (d < 16) ? powf(1.0f / 1024.0f, (float)d * (1.0f / 15.0f)) : 0.0f;
  float th = (float)t * fi;
  float c = cosf(th), s = sinf(th);
  float qr0 =  q0 * c + q1 * s, qr1 = -q0 * s + q1 * c;
  float kr0 =  k0 * c + k1 * s, kr1 = -k0 * s + k1 * c;

  float l0 = lam[0], l1 = lam[1];
  size_t veOff = (size_t)t * (HH_ * DH_) + h * DH_;
  float vb0 = l0 * v0 + l1 * ve[veOff + d];
  float vb1 = l0 * v1 + l1 * ve[veOff + d + 32];

  const int bh = b * HH_ + h;
  size_t qOff = ((size_t)bh * TT + t) * DH_;
  qb[qOff + d] = (bf16)qr0; qb[qOff + d + 32] = (bf16)qr1;
  kb[qOff + d] = (bf16)kr0; kb[qOff + d + 32] = (bf16)kr1;
  vT[((size_t)bh * DH_ + d) * TT + t]      = (bf16)vb0;
  vT[((size_t)bh * DH_ + d + 32) * TT + t] = (bf16)vb1;
}

// ---------------------------------------------------------------------------
// Flash attention. grid = (8, B*H), block = 256 (8 waves).
// Wave owns a 16-query tile; iterates causal 32-key blocks with online
// softmax. Scores & O live in WMMA f32 C-fragments; P converted to an A
// fragment via a per-wave LDS slab.
// ---------------------------------------------------------------------------
__global__ __launch_bounds__(256) void flash_attn(
    const bf16* __restrict__ qb, const bf16* __restrict__ kb,
    const bf16* __restrict__ vT, bf16* __restrict__ yb) {
  __shared__ bf16 p_lds[8][16 * 32];

  const int wave = threadIdx.x >> 5;
  const int lane = threadIdx.x & 31;
  const int m16  = lane & 15;
  const int hh   = lane >> 4;
  const int bh   = blockIdx.y;
  const int b    = bh >> 4;
  const int h    = bh & 15;
  const int qt0  = (blockIdx.x * 8 + wave) * 16;

  const bf16* qbase = qb + (size_t)bh * TT * DH_;
  const bf16* kbase = kb + (size_t)bh * TT * DH_;
  const bf16* vbase = vT + (size_t)bh * DH_ * TT;

  // Q fragments for both K-steps (dh 0..31 / 32..63), kept in registers
  BFrag aq[2];
  {
    const bf16* qrow = qbase + (size_t)(qt0 + m16) * DH_;
    for (int kk = 0; kk < 2; ++kk) {
      aq[kk].h[0] = *(const bf16x8*)(qrow + 32 * kk + 8 * hh);
      aq[kk].h[1] = *(const bf16x8*)(qrow + 32 * kk + 16 + 8 * hh);
    }
  }

  float mstate[8], lstate[8];
  f32x8 accO[4];
  for (int r = 0; r < 8; ++r) { mstate[r] = -1e30f; lstate[r] = 0.0f; }
  for (int fb = 0; fb < 4; ++fb) accO[fb] = zero8();

  for (int j0 = 0; j0 < qt0 + 16; j0 += 32) {
    // ---- S = Q K^T for two 16-key tiles ----
    f32x8 s[2];
    s[0] = zero8(); s[1] = zero8();
    for (int nb = 0; nb < 2; ++nb) {
      const bf16* krow = kbase + (size_t)(j0 + nb * 16 + m16) * DH_;
      for (int kk = 0; kk < 2; ++kk) {
        BFrag bk;
        bk.h[0] = *(const bf16x8*)(krow + 32 * kk + 8 * hh);
        bk.h[1] = *(const bf16x8*)(krow + 32 * kk + 16 + 8 * hh);
        s[nb] = wmma_bf16(aq[kk].v, bk.v, s[nb]);
      }
    }
    // ---- scale + causal mask ----
    for (int nb = 0; nb < 2; ++nb) {
      int n = j0 + nb * 16 + m16;
      for (int r = 0; r < 8; ++r) {
        int qi = qt0 + r + 8 * hh;
        float val = s[nb][r] * ATTN_SCALE;
        s[nb][r] = (n <= qi) ? val : -1e30f;
      }
    }
    // ---- online softmax (row reductions across 16-lane halves) ----
    float alpha[8];
    for (int r = 0; r < 8; ++r) {
      float mx = fmaxf(s[0][r], s[1][r]);
      for (int msk = 8; msk >= 1; msk >>= 1) mx = fmaxf(mx, __shfl_xor(mx, msk));
      float mn = fmaxf(mstate[r], mx);
      alpha[r] = expf(mstate[r] - mn);
      mstate[r] = mn;
      float p0 = expf(s[0][r] - mn);
      float p1 = expf(s[1][r] - mn);
      s[0][r] = p0; s[1][r] = p1;
      float ls = p0 + p1;
      for (int msk = 8; msk >= 1; msk >>= 1) ls += __shfl_xor(ls, msk);
      lstate[r] = lstate[r] * alpha[r] + ls;
    }
    for (int fb = 0; fb < 4; ++fb)
      for (int r = 0; r < 8; ++r) accO[fb][r] *= alpha[r];

    // ---- P: C-layout -> A-layout via per-wave LDS slab ----
    bf16* pl = p_lds[wave];
    for (int nb = 0; nb < 2; ++nb)
      for (int r = 0; r < 8; ++r)
        pl[(r + 8 * hh) * 32 + nb * 16 + m16] = (bf16)s[nb][r];
    asm volatile("s_wait_dscnt 0x0" ::: "memory");

    BFrag ap;
    const bf16* prow = pl + m16 * 32;
    ap.h[0] = *(const bf16x8*)(prow + 8 * hh);
    ap.h[1] = *(const bf16x8*)(prow + 16 + 8 * hh);

    // ---- O += P V (V^T layout gives contiguous key runs) ----
    for (int fb = 0; fb < 4; ++fb) {
      const bf16* vrow = vbase + (size_t)(fb * 16 + m16) * TT + j0 + 8 * hh;
      BFrag bv;
      bv.h[0] = *(const bf16x8*)(vrow);
      bv.h[1] = *(const bf16x8*)(vrow + 16);
      accO[fb] = wmma_bf16(ap.v, bv.v, accO[fb]);
    }
  }

  // ---- normalize + store y as bf16 [b][t][h*DH + dh] ----
  for (int r = 0; r < 8; ++r) {
    float inv = 1.0f / lstate[r];
    int t = qt0 + r + 8 * hh;
    size_t yoff = ((size_t)(b * TT + t)) * (HH_ * DH_) + h * DH_;
    for (int fb = 0; fb < 4; ++fb)
      yb[yoff + fb * 16 + m16] = (bf16)(accO[fb][r] * inv);
  }
}

// ---------------------------------------------------------------------------
// Host-side launcher
// ---------------------------------------------------------------------------
extern "C" void kernel_launch(void* const* d_in, const int* in_sizes, int n_in,
                              void* d_out, int out_size, void* d_ws, size_t ws_size,
                              hipStream_t stream) {
  const float* x   = (const float*)d_in[0];  // (B,T,D)
  const float* w   = (const float*)d_in[1];  // (4, H*DH, D)
  const float* ve  = (const float*)d_in[2];  // (4096, H*DH)
  const float* lam = (const float*)d_in[3];  // (2,)

  char* ws = (char*)d_ws;
  size_t off = 0;
  auto alloc = [&](size_t bytes) {
    void* p = ws + off;
    off += (bytes + 255) & ~(size_t)255;
    return p;
  };
  bf16*  xb    = (bf16*) alloc((size_t)NTOK * DD * 2);        // 8 MiB
  bf16*  wqkvb = (bf16*) alloc((size_t)NQKV * DD * 2);        // 6 MiB
  bf16*  wob   = (bf16*) alloc((size_t)DD * DD * 2);          // 2 MiB
  float* qkvf  = (float*)alloc((size_t)NTOK * NQKV * 4);      // 48 MiB
  bf16*  qb    = (bf16*) alloc((size_t)BB * HH_ * TT * DH_ * 2); // 8 MiB
  bf16*  kb    = (bf16*) alloc((size_t)BB * HH_ * TT * DH_ * 2); // 8 MiB
  bf16*  vT    = (bf16*) alloc((size_t)BB * HH_ * DH_ * TT * 2); // 8 MiB
  bf16*  yb    = (bf16*) alloc((size_t)NTOK * DD * 2);        // 8 MiB

  // 1) precision conversion
  cvt_f32_bf16<<<2048, 256, 0, stream>>>(x, xb, NTOK * DD);
  cvt_f32_bf16<<<2048, 256, 0, stream>>>(w, wqkvb, NQKV * DD);
  cvt_f32_bf16<<<2048, 256, 0, stream>>>(w + (size_t)3 * DD * DD, wob, DD * DD);

  // 2) qkv = x @ w_qkv^T    (M=4096, N=3072, K=1024)
  gemm_bf16_nt<<<dim3(NQKV / 64, NTOK / 128), 256, 0, stream>>>(
      xb, wqkvb, qkvf, NTOK, NQKV, DD);

  // 3) rms-norm + rotary + value-embed blend
  fuse_norm_rope_v<<<(BB * TT * HH_) / 8, 256, 0, stream>>>(
      qkvf, ve, lam, qb, kb, vT);

  // 4) flash attention
  flash_attn<<<dim3(8, BB * HH_), 256, 0, stream>>>(qb, kb, vT, yb);

  // 5) out = y @ w_o^T      (M=4096, N=1024, K=1024)
  gemm_bf16_nt<<<dim3(DD / 64, NTOK / 128), 256, 0, stream>>>(
      yb, wob, (float*)d_out, NTOK, DD, DD);
}
